// MatchModule_47296179863841
// MI455X (gfx1250) — matrix operand order repaired and checked by reading, not compiled, not run on hardware
//
#include <hip/hip_runtime.h>
#include <hip/hip_bf16.h>
#include <math.h>

// ---------------- problem constants ----------------
#define B_   16
#define K_   256
#define L_   16
#define T_   64
#define HD_  128
#define H_   4
#define DK_  32
#define BK_  (B_ * K_)     // 4096
#define BL_  (B_ * L_)     // 256
#define BLK_ (BL_ * K_)    // 65536
#define BLT_ (BL_ * T_)    // 16384

typedef _Float16 f16;
typedef __attribute__((ext_vector_type(16))) _Float16 v16h;
typedef __attribute__((ext_vector_type(8)))  _Float16 v8h;
typedef __attribute__((ext_vector_type(8)))  float    v8f;

#define BN_SCALE 0.999995f             /* 1/sqrt(1+1e-5) */
#define ATT_SCALE 0.17677669529663687f /* 1/sqrt(32) */

__device__ __forceinline__ v8f wmma_f16(v16h a, v16h b, v8f c) {
  // v_wmma_f32_16x16x32_f16: (neg_a, A, neg_b, B, c_mod, C, reuse_a, reuse_b)
  return __builtin_amdgcn_wmma_f32_16x16x32_f16(false, a, false, b, (short)0, c,
                                                false, false);
}

// A fragment: 16x32 f16, rows r0..r0+15, K = kc..kc+31, loaded from fp32 [*, 128]
__device__ __forceinline__ v16h load_a_f32(const float* __restrict__ A,
                                           int r0, int kc, int lane) {
  int m = r0 + (lane & 15);
  const float* p = A + (size_t)m * HD_ + kc + ((lane & 16) ? 16 : 0);
  v16h a;
#pragma unroll
  for (int s = 0; s < 8; ++s)  a[s] = (f16)p[s];
#pragma unroll
  for (int s = 8; s < 16; ++s) a[s] = (f16)p[s];
  return a;
}

// ---------------- kernel: pack weight W[in][out] -> f16 Wt[out][in] ----------------
__global__ void wt_pack_kernel(const float* __restrict__ W, f16* __restrict__ Wt) {
  int idx = blockIdx.x * 256 + threadIdx.x;   // 128*128
  int k = idx & 127, n = idx >> 7;
  Wt[(size_t)n * 128 + k] = (f16)W[(size_t)k * 128 + n];
}

// ---------------- kernel: centers + bbox-meta ----------------
// mb[row*12] = {center(3), c0(3), c7(3), c7-c0(3)}
__global__ void centers_mb_kernel(const float* __restrict__ corner, float* __restrict__ mb) {
  int idx = blockIdx.x * 256 + threadIdx.x;
  if (idx >= BK_) return;
  const float* c = corner + (size_t)idx * 24;
  float* o = mb + (size_t)idx * 12;
#pragma unroll
  for (int d = 0; d < 3; ++d) {
    float s = 0.f;
#pragma unroll
    for (int p = 0; p < 8; ++p) s += c[p * 3 + d];
    o[d] = s * 0.125f;
    float c0 = c[d], c7 = c[21 + d];
    o[3 + d] = c0;
    o[6 + d] = c7;
    o[9 + d] = c7 - c0;
  }
}

// ---------------- kernel: safc geometry MLP -> dist_weights ----------------
__global__ void safc_kernel(const float* __restrict__ mb,
                            const float* __restrict__ w1, const float* __restrict__ b1,
                            const float* __restrict__ g1, const float* __restrict__ bb1,
                            const float* __restrict__ w2, const float* __restrict__ b2,
                            const float* __restrict__ g2, const float* __restrict__ bb2,
                            const float* __restrict__ w3, const float* __restrict__ b3,
                            float* __restrict__ distw) {
  __shared__ float sw1[128], sb1[32], sg1[32], sbb1[32];
  __shared__ float sw2[1024], sb2[32], sg2[32], sbb2[32];
  __shared__ float sw3[128], sb3[4];
  int tid = threadIdx.x;
  for (int t = tid; t < 128; t += 256) { sw1[t] = w1[t]; sw3[t] = w3[t]; }
  for (int t = tid; t < 1024; t += 256) sw2[t] = w2[t];
  if (tid < 32) {
    sb1[tid] = b1[tid]; sg1[tid] = g1[tid]; sbb1[tid] = bb1[tid];
    sb2[tid] = b2[tid]; sg2[tid] = g2[tid]; sbb2[tid] = bb2[tid];
  }
  if (tid < 4) sb3[tid] = b3[tid];
  __syncthreads();

  int idx = blockIdx.x * 256 + tid;       // < B*K*K
  int jj = idx % K_;
  int ii = (idx / K_) % K_;
  int bb = idx / (K_ * K_);
  const float* cj = mb + (size_t)(bb * K_ + jj) * 12;
  const float* ci = mb + (size_t)(bb * K_ + ii) * 12;
  float wx = cj[0] - ci[0], wy = cj[1] - ci[1], wz = cj[2] - ci[2];
  float wd = sqrtf(wx * wx + wy * wy + wz * wz);

  float h[32];
#pragma unroll
  for (int o = 0; o < 32; ++o)
    h[o] = fmaxf(0.f, wx * sw1[o] + wy * sw1[32 + o] + wz * sw1[64 + o] + wd * sw1[96 + o] + sb1[o]);
  float m = 0.f;
#pragma unroll
  for (int o = 0; o < 32; ++o) m += h[o];
  m *= (1.f / 32.f);
  float var = 0.f;
#pragma unroll
  for (int o = 0; o < 32; ++o) { float d = h[o] - m; var += d * d; }
  float rs = rsqrtf(var * (1.f / 32.f) + 1e-5f);
#pragma unroll
  for (int o = 0; o < 32; ++o) h[o] = (h[o] - m) * rs * sg1[o] + sbb1[o];

  float h2[32];
#pragma unroll
  for (int o = 0; o < 32; ++o) {
    float s = sb2[o];
#pragma unroll
    for (int c = 0; c < 32; ++c) s += h[c] * sw2[c * 32 + o];
    h2[o] = fmaxf(0.f, s);
  }
  m = 0.f;
#pragma unroll
  for (int o = 0; o < 32; ++o) m += h2[o];
  m *= (1.f / 32.f);
  var = 0.f;
#pragma unroll
  for (int o = 0; o < 32; ++o) { float d = h2[o] - m; var += d * d; }
  rs = rsqrtf(var * (1.f / 32.f) + 1e-5f);
#pragma unroll
  for (int o = 0; o < 32; ++o) h2[o] = (h2[o] - m) * rs * sg2[o] + sbb2[o];

#pragma unroll
  for (int hh = 0; hh < 4; ++hh) {
    float s = sb3[hh];
#pragma unroll
    for (int c = 0; c < 32; ++c) s += h2[c] * sw3[c * 4 + hh];
    distw[(((size_t)bb * H_ + hh) * K_ + ii) * K_ + jj] = s;
  }
}

// ---------------- kernel: bbox embedding (12 -> 128) ----------------
__global__ void bboxemb_kernel(const float* __restrict__ mb, const float* __restrict__ w,
                               const float* __restrict__ b, float* __restrict__ out) {
  size_t idx = (size_t)blockIdx.x * 256 + threadIdx.x;  // BK*128
  int n = (int)(idx & 127);
  size_t row = idx >> 7;
  const float* mrow = mb + row * 12;
  float s = b[n];
#pragma unroll
  for (int c = 0; c < 12; ++c) s += mrow[c] * w[c * 128 + n];
  out[idx] = s;
}

// ---------------- kernel: generic WMMA GEMM, N=128, packed f16 weights ----------------
// omode 0: f32 out, +bias (+resid)
// omode 1: f32 out, +bias, BN(g,b), PReLU(alpha)
// omode 2: f16 row-major out, +bias            (Q / K projections)
// omode 3: f16 transposed out [n][m], +bias    (V projection)
__global__ void gemm128_kernel(const float* __restrict__ A, const f16* __restrict__ Wt,
                               const float* __restrict__ bias, const float* __restrict__ resid,
                               const float* __restrict__ bng, const float* __restrict__ bnb,
                               const float* __restrict__ alpha, int alpha_per_ch,
                               void* __restrict__ Cv, int M, int omode) {
  int wave = threadIdx.x >> 5;
  int lane = threadIdx.x & 31;
  int r0 = blockIdx.x * 32 + (wave >> 2) * 16;
  int n0 = blockIdx.y * 64 + (wave & 3) * 16;

  // speculative prefetch of this wave's A rows (global_prefetch_b8)
  __builtin_prefetch(A + (size_t)(r0 + (lane & 15)) * HD_, 0, 3);

  v8f acc = {};
#pragma unroll
  for (int kc = 0; kc < 128; kc += 32) {
    v16h a = load_a_f32(A, r0, kc, lane);
    v16h b;
    {
      const f16* p = Wt + (size_t)(n0 + (lane & 15)) * 128 + kc + ((lane & 16) ? 16 : 0);
#pragma unroll
      for (int s = 0; s < 16; ++s) b[s] = p[s];   // 2x contiguous b128
    }
    acc = wmma_f16(a, b, acc);
  }

  int n = n0 + (lane & 15);
  int mrow0 = r0 + ((lane & 16) ? 8 : 0);
  float bs = bias[n];

  if (omode == 3) {            // f16 transposed: one contiguous 8xf16 store per lane
    v8h o;
#pragma unroll
    for (int r = 0; r < 8; ++r) o[r] = (f16)(acc[r] + bs);
    *(v8h*)((f16*)Cv + (size_t)n * M + mrow0) = o;
    return;
  }
  if (omode == 2) {            // f16 row-major
    f16* Ch = (f16*)Cv;
#pragma unroll
    for (int r = 0; r < 8; ++r)
      Ch[(size_t)(mrow0 + r) * HD_ + n] = (f16)(acc[r] + bs);
    return;
  }

  float* C = (float*)Cv;
  float bn_g = 0.f, bn_b = 0.f, al = 0.f;
  if (omode == 1) {
    bn_g = bng[n]; bn_b = bnb[n];
    al = alpha_per_ch ? alpha[n] : alpha[0];
  }
#pragma unroll
  for (int r = 0; r < 8; ++r) {
    int mrow = mrow0 + r;
    float v = acc[r] + bs;
    if (omode == 1) {
      v = v * BN_SCALE * bn_g + bn_b;
      v = (v > 0.f) ? v : al * v;
    }
    if (resid) v += resid[(size_t)mrow * HD_ + n];
    C[(size_t)mrow * HD_ + n] = v;
  }
}

// ---------------- kernel: LayerNorm over 128 ----------------
__global__ void ln128_kernel(const float* __restrict__ X, const float* __restrict__ g,
                             const float* __restrict__ b, float* __restrict__ Y, int M) {
  int row = blockIdx.x * 8 + (threadIdx.x >> 5);
  int lane = threadIdx.x & 31;
  if (row >= M) return;
  const float* x = X + (size_t)row * 128;
  float v[4];
#pragma unroll
  for (int i = 0; i < 4; ++i) v[i] = x[lane + 32 * i];
  float s = v[0] + v[1] + v[2] + v[3];
#pragma unroll
  for (int off = 16; off >= 1; off >>= 1) s += __shfl_xor(s, off, 32);
  float mean = s * (1.f / 128.f);
  float q = 0.f;
#pragma unroll
  for (int i = 0; i < 4; ++i) { float d = v[i] - mean; q += d * d; }
#pragma unroll
  for (int off = 16; off >= 1; off >>= 1) q += __shfl_xor(q, off, 32);
  float rstd = rsqrtf(q * (1.f / 128.f) + 1e-5f);
#pragma unroll
  for (int i = 0; i < 4; ++i) {
    int n = lane + 32 * i;
    Y[(size_t)row * 128 + n] = (v[i] - mean) * rstd * g[n] + b[n];
  }
}

// ---------------- kernel: fused attention ----------------
// Q: f16 [Mq][128] row-major; Kb: f16 [Mk][128] row-major; Vt: f16 [128][Mk] transposed.
// One wave per (16 q-rows, head, batch). Scores staged in LDS (<=16x256).
__global__ void attention_kernel(const f16* __restrict__ Q, const f16* __restrict__ Kb,
                                 const f16* __restrict__ Vt, int Mkv,
                                 float* __restrict__ O, int Nq, int Nk,
                                 const float* __restrict__ bias, int bias_div,
                                 const unsigned char* __restrict__ mask) {
  __shared__ float s_s[16 * 256];
  __shared__ _Float16 s_p[16 * 256];
  int lane = threadIdx.x & 31;
  int qt = blockIdx.x, h = blockIdx.y, b = blockIdx.z;
  int qbase = b * Nq, kvbase = b * Nk;

  // Q fragment (A: 16x32) -- contiguous f16
  v16h qa;
  {
    int m = qbase + qt * 16 + (lane & 15);
    const f16* p = Q + (size_t)m * HD_ + h * DK_ + ((lane & 16) ? 16 : 0);
#pragma unroll
    for (int s = 0; s < 8; ++s)  qa[s] = p[s];
#pragma unroll
    for (int s = 8; s < 16; ++s) qa[s] = p[s];
  }

  size_t boff = 0;
  if (bias) boff = (size_t)((b / bias_div) * H_ + h) * Nq * Nk;

  for (int nt = 0; nt < Nk / 16; ++nt) {
    // B = K^T fragment: contiguous 16 halves per lane
    v16h kb;
    {
      int n = kvbase + nt * 16 + (lane & 15);
      const f16* p = Kb + (size_t)n * HD_ + h * DK_ + ((lane & 16) ? 16 : 0);
#pragma unroll
      for (int s = 0; s < 16; ++s) kb[s] = p[s];
    }
    v8f c = {};
    c = wmma_f16(qa, kb, c);
    int col = nt * 16 + (lane & 15);
    bool mk = mask ? (mask[(size_t)b * Nk + col] != 0) : false;
#pragma unroll
    for (int r = 0; r < 8; ++r) {
      int rr = r + ((lane & 16) ? 8 : 0);
      float v = c[r] * ATT_SCALE;
      if (bias) v += bias[boff + (size_t)(qt * 16 + rr) * Nk + col];
      if (mk) v = -INFINITY;
      s_s[rr * 256 + col] = v;
    }
  }
  __syncthreads();

  if (lane < 16) {
    float* row = s_s + lane * 256;
    float mx = -INFINITY;
    for (int k = 0; k < Nk; ++k) mx = fmaxf(mx, row[k]);
    float sum = 0.f;
    for (int k = 0; k < Nk; ++k) { float e = __expf(row[k] - mx); row[k] = e; sum += e; }
    float inv = 1.f / sum;
    _Float16* prow = s_p + lane * 256;
    for (int k = 0; k < Nk; ++k) prow[k] = (f16)(row[k] * inv);
  }
  __syncthreads();

  v8f o0 = {}, o1 = {};
  for (int kc = 0; kc < Nk; kc += 32) {
    v16h pa;
    {
      const _Float16* p = s_p + (lane & 15) * 256 + kc + ((lane & 16) ? 16 : 0);
#pragma unroll
      for (int s = 0; s < 8; ++s)  pa[s] = p[s];
#pragma unroll
      for (int s = 8; s < 16; ++s) pa[s] = p[s];
    }
    // V fragments from transposed f16 V: contiguous per lane
    int kr = kvbase + kc + ((lane & 16) ? 16 : 0);
    const f16* vp0 = Vt + (size_t)(h * DK_ + (lane & 15)) * Mkv + kr;
    const f16* vp1 = Vt + (size_t)(h * DK_ + 16 + (lane & 15)) * Mkv + kr;
    v16h vb0, vb1;
#pragma unroll
    for (int s = 0; s < 16; ++s) { vb0[s] = vp0[s]; vb1[s] = vp1[s]; }
    o0 = wmma_f16(pa, vb0, o0);
    o1 = wmma_f16(pa, vb1, o1);
  }

  int m0 = qbase + qt * 16 + ((lane & 16) ? 8 : 0);
  int d = h * DK_ + (lane & 15);
#pragma unroll
  for (int r = 0; r < 8; ++r) {
    O[(size_t)(m0 + r) * HD_ + d] = o0[r];
    O[(size_t)(m0 + r) * HD_ + d + 16] = o1[r];
  }
}

// ---------------- kernel: broadcast features over L ----------------
__global__ void broadcast_kernel(const float* __restrict__ src, float* __restrict__ dst) {
  size_t idx = (size_t)blockIdx.x * 256 + threadIdx.x;  // BLK*128
  int c = (int)(idx & 127);
  size_t row = idx >> 7;            // bl*K + k
  int k = (int)(row % K_);
  int bl = (int)(row / K_);
  int b = bl / L_;
  dst[idx] = src[((size_t)(b * K_ + k)) * HD_ + c];
}

// ---------------- kernel: final 128 -> 1 head ----------------
__global__ void head_kernel(const float* __restrict__ A, const float* __restrict__ w3,
                            const float* __restrict__ b3, float* __restrict__ out, int M) {
  int row = blockIdx.x * 8 + (threadIdx.x >> 5);
  int lane = threadIdx.x & 31;
  if (row >= M) return;
  float s = 0.f;
#pragma unroll
  for (int i = 0; i < 4; ++i) {
    int c = lane + 32 * i;
    s += A[(size_t)row * 128 + c] * w3[c];
  }
#pragma unroll
  for (int off = 16; off >= 1; off >>= 1) s += __shfl_xor(s, off, 32);
  if (lane == 0) out[row] = s + b3[0];
}

// ---------------- host orchestration ----------------
// Input flattening assumption (insertion order, recursive):
//  0 pred_bbox_corner  1 pred_bbox_feature  2 lang_fea  3 attention_mask  4 len_nun_max
//  fc:    5 w1  6 b1  7 bn1_g  8 bn1_b  9 prelu  10 w2  11 b2
//  safc: 12 w1 13 b1 14 ln1_g 15 ln1_b 16 w2 17 b2 18 ln2_g 19 ln2_b 20 w3 21 b3
//  bbox: 22 w 23 b
//  sa0: 24.. sa1: 34.. ca0: 44.. ca1: 54..  (wq,bq,wk,bk,wv,bv,wo,bo,ln_g,ln_b)
//  match: 64 w1 65 b1 66 g1 67 b1 68 p1 69 w2 70 b2 71 g2 72 b2 73 p2 74 w3 75 b3
extern "C" void kernel_launch(void* const* d_in, const int* in_sizes, int n_in,
                              void* d_out, int out_size, void* d_ws, size_t ws_size,
                              hipStream_t stream) {
  auto F = [&](int i) { return (const float*)d_in[i]; };
  const float* corner = F(0);
  const float* bfeat  = F(1);
  const float* lang   = F(2);
  const unsigned char* amask = (const unsigned char*)d_in[3];

  char* wsb = (char*)d_ws;
  size_t off = 0;
  auto alloc = [&](size_t bytes) {
    void* p = wsb + off;
    off += (bytes + 255) & ~(size_t)255;
    return p;
  };
  float* mb    = (float*)alloc((size_t)BK_ * 12 * 4);
  float* distw = (float*)alloc((size_t)B_ * H_ * K_ * K_ * 4);
  float* feat  = (float*)alloc((size_t)BK_ * HD_ * 4);
  float* bbemb = (float*)alloc((size_t)BK_ * HD_ * 4);
  float* t1    = (float*)alloc((size_t)BLK_ * HD_ * 4);
  float* t2    = (float*)alloc((size_t)BLK_ * HD_ * 4);
  float* ao    = (float*)alloc((size_t)BLK_ * HD_ * 4);
  float* f1    = (float*)alloc((size_t)BLK_ * HD_ * 4);
  float* f2    = (float*)alloc((size_t)BLK_ * HD_ * 4);
  f16*   Qh    = (f16*)alloc((size_t)BLK_ * HD_ * 2);
  f16*   Kh    = (f16*)alloc((size_t)BLK_ * HD_ * 2);
  f16*   Vth   = (f16*)alloc((size_t)BLK_ * HD_ * 2);
  f16*   wpack = (f16*)alloc((size_t)20 * 16384 * 2);
  float* lang_f = nullptr;  // lang used directly (f32 A input)
  (void)lang_f;
  if (ws_size < off) return;  // deterministic guard

  // ---- pack all 128x128 weights once: f16, transposed to [out][in] ----
  const int widx[20] = {5, 10, 24, 26, 28, 30, 34, 36, 38, 40,
                        44, 46, 48, 50, 54, 56, 58, 60, 64, 69};
  for (int i = 0; i < 20; ++i)
    wt_pack_kernel<<<64, 256, 0, stream>>>(F(widx[i]), wpack + (size_t)i * 16384);
  auto WP = [&](int inIdx) -> const f16* {
    for (int s = 0; s < 20; ++s)
      if (widx[s] == inIdx) return wpack + (size_t)s * 16384;
    return wpack;
  };

  auto gemm = [&](const float* A, const f16* Wt, const float* bias, const float* resid,
                  const float* g, const float* bb, const float* al, int apc,
                  void* C, int M, int omode) {
    gemm128_kernel<<<dim3(M / 32, 2), 256, 0, stream>>>(A, Wt, bias, resid, g, bb, al, apc,
                                                        C, M, omode);
  };
  auto mha = [&](int base, const float* xq, int nb, int Nq, const float* xkv, int Nk,
                 const float* bias, int bias_div, const unsigned char* mask, float* outp) {
    int Mq = nb * Nq, Mk = nb * Nk;
    gemm(xq,  WP(base + 0), F(base + 1), nullptr, nullptr, nullptr, nullptr, 0, Qh,  Mq, 2);
    gemm(xkv, WP(base + 2), F(base + 3), nullptr, nullptr, nullptr, nullptr, 0, Kh,  Mk, 2);
    gemm(xkv, WP(base + 4), F(base + 5), nullptr, nullptr, nullptr, nullptr, 0, Vth, Mk, 3);
    attention_kernel<<<dim3(Nq / 16, H_, nb), 32, 0, stream>>>(Qh, Kh, Vth, Mk, ao, Nq, Nk,
                                                               bias, bias_div, mask);
    gemm(ao, WP(base + 6), F(base + 7), xq, nullptr, nullptr, nullptr, 0, t1, Mq, 0);
    ln128_kernel<<<Mq / 8, 256, 0, stream>>>(t1, F(base + 8), F(base + 9), outp, Mq);
  };

  // geometry bias
  centers_mb_kernel<<<BK_ / 256, 256, 0, stream>>>(corner, mb);
  safc_kernel<<<(B_ * K_ * K_) / 256, 256, 0, stream>>>(mb, F(12), F(13), F(14), F(15),
                                                        F(16), F(17), F(18), F(19), F(20), F(21),
                                                        distw);
  // feature MLP + bbox embedding
  bboxemb_kernel<<<(BK_ * HD_) / 256, 256, 0, stream>>>(mb, F(22), F(23), bbemb);
  gemm(bfeat, WP(5), F(6), nullptr, F(7), F(8), F(9), 1, t1, BK_, 1);        // fc1+bn+prelu
  gemm(t1, WP(10), F(11), bbemb, nullptr, nullptr, nullptr, 0, feat, BK_, 0); // fc2 + bbox resid
  // sa0 (B=16)
  mha(24, feat, B_, K_, feat, K_, distw, 1, nullptr, t2);
  // broadcast to B*L
  broadcast_kernel<<<((size_t)BLK_ * HD_) / 256, 256, 0, stream>>>(t2, f1);
  // ca0, sa1, ca1 (B*L = 256)
  mha(44, f1, BL_, K_, lang, T_, nullptr, 1, amask, f2);
  mha(34, f2, BL_, K_, f2, K_, distw, L_, nullptr, f1);
  mha(54, f1, BL_, K_, lang, T_, nullptr, 1, amask, f2);
  // match MLP
  gemm(f2, WP(64), F(65), nullptr, F(66), F(67), F(68), 0, t1, BLK_, 1);
  gemm(t1, WP(69), F(70), nullptr, F(71), F(72), F(73), 0, t2, BLK_, 1);
  head_kernel<<<BLK_ / 8, 256, 0, stream>>>(t2, F(74), F(75), (float*)d_out, BLK_);
}
